// Head_65171833749740
// MI455X (gfx1250) — compile-verified, matrix operand order checked by reading, not run on hardware
//
#include <hip/hip_runtime.h>
#include <hip/hip_bf16.h>

typedef __attribute__((ext_vector_type(16))) __bf16 v16bf;
typedef __attribute__((ext_vector_type(8)))  __bf16 v8bf;
typedef __attribute__((ext_vector_type(2)))  __bf16 v2bf;
typedef __attribute__((ext_vector_type(8)))  float  v8f;
typedef __attribute__((ext_vector_type(4)))  float  v4f;

#define WMMA_BF16(a,b,c) \
  __builtin_amdgcn_wmma_f32_16x16x32_bf16(false,(a),false,(b),(short)0,(c),false,false)

// combine two contiguous 8-element halves into one A/B fragment
#define FRAG16(lo,hi) __builtin_shufflevector((lo),(hi),0,1,2,3,4,5,6,7,8,9,10,11,12,13,14,15)

static constexpr int Bsz = 8;
static constexpr int T   = 4096;
static constexpr int E   = 1024;
static constexpr int H   = 128;

// ---------------------------------------------------------------------------
// Kernel 0: Wt[s][n][e] = bf16(W_s[e][n])  -- transposed bf16 weights so that
// projection B-fragments become contiguous b128 loads.
// ---------------------------------------------------------------------------
__global__ __launch_bounds__(256)
void wprep_kernel(const float* __restrict__ Wq,
                  const float* __restrict__ Wk,
                  const float* __restrict__ Wv,
                  __bf16* __restrict__ wt) {
  const int idx = blockIdx.x * blockDim.x + threadIdx.x;
  const int per = E * H;
  if (idx >= 3 * per) return;
  const int s = idx / per;
  const int r = idx % per;
  const int e = r / H;
  const int n = r % H;
  const float* W = (s == 0) ? Wq : (s == 1) ? Wk : Wv;
  wt[(size_t)s * per + (size_t)n * E + e] = (__bf16)W[(size_t)e * H + n];
}

// ---------------------------------------------------------------------------
// Kernel 1: q/k = x @ W (row-major bf16), v stored transposed vT[B][H][T].
// grid = (B*T/64, 3), block = 128 (4 waves). Each wave: 16 rows x 128 cols.
// ---------------------------------------------------------------------------
__global__ __launch_bounds__(128)
void qkv_proj_kernel(const float* __restrict__ x,
                     const __bf16* __restrict__ wt,   // [3][H][E] transposed bf16
                     __bf16* __restrict__ qo,
                     __bf16* __restrict__ ko,
                     __bf16* __restrict__ vt) {       // [B][H][T]
  const int lane = threadIdx.x & 31;
  const int wave = threadIdx.x >> 5;
  const int half = lane >> 4;
  const int l16  = lane & 15;
  const int row0 = blockIdx.x * 64 + wave * 16;
  const int sel  = blockIdx.y;

  const __bf16* Wt = wt + (size_t)sel * E * H;

  v8f acc[8] = {};

  const float* xrow = x + (size_t)(row0 + l16) * E;
  for (int e0 = 0; e0 < E; e0 += 32) {
    // ---- batch-issue ALL loads for this K chunk first (MLP) ----
    const v4f* xp4 = reinterpret_cast<const v4f*>(xrow + e0 + (half ? 8 : 0));
    const v4f x0 = xp4[0], x1 = xp4[1], x2 = xp4[4], x3 = xp4[5];

    v8bf wlo[8], whi[8];
    #pragma unroll
    for (int c = 0; c < 8; ++c) {
      const v8bf* wp8 = reinterpret_cast<const v8bf*>(
          Wt + (size_t)(c * 16 + l16) * E + e0 + (half ? 8 : 0));
      wlo[c] = wp8[0];
      whi[c] = wp8[2];
    }
    __builtin_prefetch(xrow + e0 + 32, 0, 0);

    // ---- convert A fragment ----
    v16bf a;
    #pragma unroll
    for (int i = 0; i < 4; ++i) {
      a[i]      = (__bf16)x0[i];
      a[4 + i]  = (__bf16)x1[i];
      a[8 + i]  = (__bf16)x2[i];
      a[12 + i] = (__bf16)x3[i];
    }

    // ---- back-to-back WMMA chain ----
    #pragma unroll
    for (int c = 0; c < 8; ++c)
      acc[c] = WMMA_BF16(a, FRAG16(wlo[c], whi[c]), acc[c]);
  }

  // ---- store ----
  if (sel == 2) {
    // v transposed: vT[b][h][t]
    #pragma unroll
    for (int c = 0; c < 8; ++c) {
      #pragma unroll
      for (int r = 0; r < 8; ++r) {
        const int row_g = row0 + (half ? 8 : 0) + r;
        const int bidx  = row_g >> 12;        // / T
        const int t     = row_g & (T - 1);
        vt[(size_t)bidx * H * T + (size_t)(c * 16 + l16) * T + t] = (__bf16)acc[c][r];
      }
    }
  } else {
    __bf16* out = (sel == 0) ? qo : ko;
    #pragma unroll
    for (int c = 0; c < 8; ++c) {
      #pragma unroll
      for (int r = 0; r < 8; ++r) {
        const int m = (half ? 8 : 0) + r;
        out[(size_t)(row0 + m) * H + c * 16 + l16] = (__bf16)acc[c][r];
      }
    }
  }
}

// ---------------------------------------------------------------------------
// Kernel 2: causal flash attention. grid = (T/64, B), block = 128 (4 waves).
// q,k row-major bf16 [B][T][H]; v transposed bf16 [B][H][T]; out f32.
// ---------------------------------------------------------------------------
__global__ __launch_bounds__(128)
void flash_attn_kernel(const __bf16* __restrict__ q,
                       const __bf16* __restrict__ k,
                       const __bf16* __restrict__ vt,
                       float* __restrict__ out) {
  __shared__ v2bf plds[4][16][16];   // per-wave P staging, packed (4 KB)

  const int lane = threadIdx.x & 31;
  const int wave = threadIdx.x >> 5;
  const int half = lane >> 4;
  const int l16  = lane & 15;
  const int b    = blockIdx.y;
  const int q0   = blockIdx.x * 64 + wave * 16;

  const __bf16* qb  = q  + (size_t)b * T * H;
  const __bf16* kb  = k  + (size_t)b * T * H;
  const __bf16* vbt = vt + (size_t)b * H * T;

  // ---- q tile as 4 A-fragments: 8 x b128 loads ----
  v16bf qa[4];
  {
    const v8bf* qp8 = reinterpret_cast<const v8bf*>(
        qb + (size_t)(q0 + l16) * H + (half ? 8 : 0));
    #pragma unroll
    for (int c4 = 0; c4 < 4; ++c4)
      qa[c4] = FRAG16(qp8[c4 * 4], qp8[c4 * 4 + 2]);
  }

  v8f   o[8] = {};
  float mrow[8], lrow[8];
  #pragma unroll
  for (int r = 0; r < 8; ++r) { mrow[r] = -__builtin_inff(); lrow[r] = 0.f; }

  const float scale = 0.03125f;        // 1024^-0.5
  const int   nkt   = (q0 + 47) >> 5;  // causal bound on 32-key tiles

  for (int kt = 0; kt < nkt; ++kt) {
    const int kb0 = kt * 32;

    // ---- batch-load k fragments (both 16-col halves of the 32-key tile) ----
    v8bf klo[2][4], khi[2][4];
    #pragma unroll
    for (int n0 = 0; n0 < 2; ++n0) {
      const v8bf* kp8 = reinterpret_cast<const v8bf*>(
          kb + (size_t)(kb0 + n0 * 16 + l16) * H + (half ? 8 : 0));
      #pragma unroll
      for (int c4 = 0; c4 < 4; ++c4) {
        klo[n0][c4] = kp8[c4 * 4];
        khi[n0][c4] = kp8[c4 * 4 + 2];
      }
    }
    // ---- batch-load v fragments for this key tile (vT: contiguous b128) ----
    v8bf vlo[8], vhi[8];
    #pragma unroll
    for (int c = 0; c < 8; ++c) {
      const v8bf* vp8 = reinterpret_cast<const v8bf*>(
          vbt + (size_t)(c * 16 + l16) * T + kb0 + (half ? 8 : 0));
      vlo[c] = vp8[0];
      vhi[c] = vp8[2];
    }

    // ---- S = q @ k^T : 8 back-to-back WMMAs ----
    v8f s[2] = {};
    #pragma unroll
    for (int n0 = 0; n0 < 2; ++n0)
      #pragma unroll
      for (int c4 = 0; c4 < 4; ++c4)
        s[n0] = WMMA_BF16(qa[c4], FRAG16(klo[n0][c4], khi[n0][c4]), s[n0]);

    // ---- online softmax (row = half*8 + r spans 16 lanes) ----
    #pragma unroll
    for (int r = 0; r < 8; ++r) {
      const int qrow = q0 + (half ? 8 : 0) + r;
      float s0 = s[0][r] * scale;
      float s1 = s[1][r] * scale;
      if (kb0 + l16      > qrow) s0 = -__builtin_inff();
      if (kb0 + 16 + l16 > qrow) s1 = -__builtin_inff();

      float mx = fmaxf(s0, s1);
      #pragma unroll
      for (int off = 8; off >= 1; off >>= 1)
        mx = fmaxf(mx, __shfl_xor(mx, off, 16));

      const float newm  = fmaxf(mrow[r], mx);
      const float alpha = __expf(mrow[r] - newm);
      mrow[r] = newm;

      s0 = __expf(s0 - newm);
      s1 = __expf(s1 - newm);
      float rs = s0 + s1;
      #pragma unroll
      for (int off = 8; off >= 1; off >>= 1)
        rs += __shfl_xor(rs, off, 16);
      lrow[r] = lrow[r] * alpha + rs;

      #pragma unroll
      for (int c = 0; c < 8; ++c) o[c][r] = o[c][r] * alpha;

      // packed dword store: pair (col, col+16) of row (half*8+r)
      v2bf p;
      p[0] = (__bf16)s0;
      p[1] = (__bf16)s1;
      plds[wave][(half ? 8 : 0) + r][l16] = p;     // ds_store_b32
    }

    // ---- P as A-fragment: 2 x ds_load_b128 + deinterleave shuffle ----
    v16bf pa;
    {
      const v16bf raw = *reinterpret_cast<const v16bf*>(
          &plds[wave][l16][half ? 8 : 0]);
      // raw = [c0,c16,c1,c17,...]; want [c0..c7, c16..c23]
      pa = __builtin_shufflevector(raw, raw,
                                   0, 2, 4, 6, 8, 10, 12, 14,
                                   1, 3, 5, 7, 9, 11, 13, 15);
    }

    // ---- O += P @ V : 8 back-to-back WMMAs ----
    #pragma unroll
    for (int c = 0; c < 8; ++c)
      o[c] = WMMA_BF16(pa, FRAG16(vlo[c], vhi[c]), o[c]);
  }

  // ---- epilogue: out = O / l, f32 ----
  float* ob = out + (size_t)b * T * H;
  #pragma unroll
  for (int r = 0; r < 8; ++r) {
    const float inv = 1.f / lrow[r];
    const int   m   = (half ? 8 : 0) + r;
    #pragma unroll
    for (int c = 0; c < 8; ++c)
      ob[(size_t)(q0 + m) * H + c * 16 + l16] = o[c][r] * inv;
  }
}

// ---------------------------------------------------------------------------
extern "C" void kernel_launch(void* const* d_in, const int* in_sizes, int n_in,
                              void* d_out, int out_size, void* d_ws, size_t ws_size,
                              hipStream_t stream) {
  const float* x  = (const float*)d_in[0];
  const float* Wk = (const float*)d_in[1];
  const float* Wq = (const float*)d_in[2];
  const float* Wv = (const float*)d_in[3];
  float* out = (float*)d_out;

  __bf16* wt   = (__bf16*)d_ws;                       // 768 KB: [3][H][E]
  __bf16* qws  = wt  + (size_t)3 * E * H;             // 8 MB
  __bf16* kws  = qws + (size_t)Bsz * T * H;           // 8 MB
  __bf16* vtws = kws + (size_t)Bsz * T * H;           // 8 MB (transposed)

  // order in wt: [0]=Wq, [1]=Wk, [2]=Wv
  wprep_kernel<<<(3 * E * H + 255) / 256, 256, 0, stream>>>(Wq, Wk, Wv, wt);

  dim3 g1(Bsz * T / 64, 3);
  qkv_proj_kernel<<<g1, 128, 0, stream>>>(x, wt, qws, kws, vtws);

  dim3 g2(T / 64, Bsz);
  flash_attn_kernel<<<g2, 128, 0, stream>>>(qws, kws, vtws, out);
}
